// DASGNNAggregator_26173530702072
// MI455X (gfx1250) — matrix operand order repaired and checked by reading, not compiled, not run on hardware
//
#include <hip/hip_runtime.h>

typedef __attribute__((ext_vector_type(16))) _Float16 v16h;
typedef __attribute__((ext_vector_type(8)))  _Float16 v8h;
typedef __attribute__((ext_vector_type(8)))  float    v8f;

#define N_NODES 20000
#define KNEI    32
#define DIN     256
#define DOUT    256
#define KSEL    16

// ---------------------------------------------------------------------------
// Kernel 1: pack f32 weights W[K=256][N=256] into f16 WMMA B-fragment layout.
// Fragment block b = ntile*8 + kstep is a 32(K)x16(N) tile stored as
// 32 lanes x 16 halves, lane-contiguous:
//   lane l  -> column N = ntile*16 + (l & 15)
//   elem e  -> K = kstep*32 + (e & 7) + 16*(e >= 8) + 8*(l >= 16)
// so each lane's 16 halves (32 B) are contiguous in the packed buffer.
// ---------------------------------------------------------------------------
__global__ void pack_weights(const float* __restrict__ W, _Float16* __restrict__ P) {
  int g      = blockIdx.x * blockDim.x + threadIdx.x;   // 65536 total
  int blk    = g >> 9;
  int within = g & 511;
  int lane   = within >> 4;
  int e      = within & 15;
  int ntile  = blk >> 3;
  int ks     = blk & 7;
  int K = ks * 32 + (e & 7) + ((e >= 8) ? 16 : 0) + ((lane >= 16) ? 8 : 0);
  int N = ntile * 16 + (lane & 15);
  P[g] = (_Float16)W[K * DOUT + N];
}

// ---------------------------------------------------------------------------
// Kernel 2: st = self_vecs(20000x256) @ self_weights(256x256), f16 WMMA,
// fp32 accumulate/output. 16 rows per block, 4 waves x 4 N-tiles.
// A fragment layout (16-bit A 16x32): lane<16 holds row M=l, K={0..7,16..23};
// lane>=16 same rows, K shifted by +8.
// ---------------------------------------------------------------------------
__global__ __launch_bounds__(128) void st_gemm(const float* __restrict__ self_vecs,
                                               const _Float16* __restrict__ Wp,
                                               float* __restrict__ st) {
  const int lane  = threadIdx.x & 31;
  const int warp  = threadIdx.x >> 5;     // 0..3
  const int mbase = blockIdx.x * 16;
  const int row   = mbase + (lane & 15);
  const int hi    = (lane >> 4) * 8;

  v8f acc[4] = {};
  for (int ks = 0; ks < 8; ++ks) {
    v16h a;
    const float* ap = self_vecs + (size_t)row * DIN + ks * 32 + hi;
#pragma unroll
    for (int i = 0; i < 8; ++i) {
      a[i]     = (_Float16)ap[i];
      a[i + 8] = (_Float16)ap[16 + i];
    }
#pragma unroll
    for (int t = 0; t < 4; ++t) {
      int ntile = warp * 4 + t;
      v16h b = *((const v16h*)Wp + (ntile * 8 + ks) * 32 + lane);
      acc[t] = __builtin_amdgcn_wmma_f32_16x16x32_f16(false, a, false, b,
                                                      (short)0, acc[t], false, false);
    }
  }
  // C/D layout: VGPR r, lane l -> M = r + 8*(l>=16), N = l&15
#pragma unroll
  for (int t = 0; t < 4; ++t) {
    int col = (warp * 4 + t) * 16 + (lane & 15);
    int r0  = (lane >> 4) * 8;
#pragma unroll
    for (int r = 0; r < 8; ++r)
      st[(size_t)(mbase + r0 + r) * DOUT + col] = acc[t][r];
  }
}

// ---------------------------------------------------------------------------
// Kernel 3: fused neighbor GEMM + attention softmax + top-16 + aggregate.
// 2 nodes per 256-thread block (8 wave32s). tileH holds A (f16) during the
// WMMA loop, then is reused to hold nt (f16) for the epilogue.
// ---------------------------------------------------------------------------
__global__ __launch_bounds__(256) void agg_kernel(const float* __restrict__ neigh_vecs,
                                                  const _Float16* __restrict__ Wnp,
                                                  const float* __restrict__ stBuf,
                                                  const float* __restrict__ attn,
                                                  float* __restrict__ out) {
  __shared__ __attribute__((aligned(32))) _Float16 tileH[2][KNEI][DIN]; // 32 KB
  __shared__ float attnLds[DOUT];
  __shared__ float stLds[2][DOUT];
  __shared__ float logitsLds[2][33];
  __shared__ float wselLds[2][KNEI];

  const int tid  = threadIdx.x;
  const int lane = tid & 31;
  const int warp = tid >> 5;
  const int n0   = blockIdx.x * 2;

  // ---- stage neighbor rows (f32 -> f16), attention vector, st rows ----
  {
    const float4* src = (const float4*)(neigh_vecs + (size_t)n0 * KNEI * DIN);
    for (int f = tid; f < 2 * KNEI * DIN / 4; f += 256) {
      float4 v = src[f];
      _Float16* dst = &tileH[0][0][0] + f * 4;
      dst[0] = (_Float16)v.x; dst[1] = (_Float16)v.y;
      dst[2] = (_Float16)v.z; dst[3] = (_Float16)v.w;
    }
    attnLds[tid]  = attn[tid];
    stLds[0][tid] = stBuf[(size_t)n0 * DOUT + tid];
    stLds[1][tid] = stBuf[(size_t)(n0 + 1) * DOUT + tid];
  }
  __syncthreads();

  // ---- WMMA: wave w owns N-tiles {2w,2w+1}; B frags reused across 2 nodes ----
  v8f acc[2][2][2] = {};                // [node][mtile][ntile]
  const int row16 = lane & 15;
  const int hi    = (lane >> 4) * 8;
  const int nt0   = warp * 2;
  for (int ks = 0; ks < 8; ++ks) {
    v16h b0 = *((const v16h*)Wnp + ((nt0 + 0) * 8 + ks) * 32 + lane);
    v16h b1 = *((const v16h*)Wnp + ((nt0 + 1) * 8 + ks) * 32 + lane);
#pragma unroll
    for (int nd = 0; nd < 2; ++nd) {
#pragma unroll
      for (int mt = 0; mt < 2; ++mt) {
        const _Float16* ap = &tileH[nd][mt * 16 + row16][ks * 32 + hi];
        v8h lo = *(const v8h*)ap;
        v8h hh = *(const v8h*)(ap + 16);
        v16h a;
#pragma unroll
        for (int i = 0; i < 8; ++i) { a[i] = lo[i]; a[i + 8] = hh[i]; }
        acc[nd][mt][0] = __builtin_amdgcn_wmma_f32_16x16x32_f16(false, a, false, b0,
                                        (short)0, acc[nd][mt][0], false, false);
        acc[nd][mt][1] = __builtin_amdgcn_wmma_f32_16x16x32_f16(false, a, false, b1,
                                        (short)0, acc[nd][mt][1], false, false);
      }
    }
  }
  __syncthreads();     // A fully consumed by all waves

  // ---- spill nt into the (now free) tileH region as f16 ----
#pragma unroll
  for (int nd = 0; nd < 2; ++nd)
#pragma unroll
    for (int mt = 0; mt < 2; ++mt)
#pragma unroll
      for (int t = 0; t < 2; ++t) {
        int col = (nt0 + t) * 16 + row16;
        int r0  = (lane >> 4) * 8;
#pragma unroll
        for (int r = 0; r < 8; ++r)
          tileH[nd][mt * 16 + r0 + r][col] = (_Float16)acc[nd][mt][t][r];
      }
  __syncthreads();

  // ---- logits: 66 rows (2 nodes x 33), one row per wave iteration ----
  for (int q = warp; q < 66; q += 8) {
    int nd = q / 33, j = q % 33;
    float sum = 0.f;
#pragma unroll
    for (int i = 0; i < 8; ++i) {
      int c   = lane + 32 * i;
      float v = (j == 0) ? stLds[nd][c] : (float)tileH[nd][j - 1][c];
      sum += v * attnLds[c];
    }
#pragma unroll
    for (int off = 16; off > 0; off >>= 1) sum += __shfl_xor(sum, off);
    if (lane == 0) logitsLds[nd][j] = fmaxf(sum, 0.f);  // relu
  }
  __syncthreads();

  // ---- softmax over 33 + top-16 selection (lowest-index tie-break) ----
  if (warp < 2) {
    int nd   = warp;
    float x  = logitsLds[nd][lane + 1];   // neighbor logit
    float xs = logitsLds[nd][0];          // self logit
    float m = x;
#pragma unroll
    for (int off = 16; off > 0; off >>= 1) m = fmaxf(m, __shfl_xor(m, off));
    m = fmaxf(m, xs);
    float e = expf(x - m);
    float Z = e;
#pragma unroll
    for (int off = 16; off > 0; off >>= 1) Z += __shfl_xor(Z, off);
    Z += expf(xs - m);
    float s = e / Z;                      // neighbor softmax score
    int rank = 0;
    for (int i = 0; i < 32; ++i) {
      float si = __shfl(s, i);
      rank += (si > s) || (si == s && i < lane);
    }
    wselLds[nd][lane] = (rank < KSEL) ? s : 0.f;
  }
  __syncthreads();

  // ---- aggregate selected neighbors and emit relu(st + agg) ----
#pragma unroll
  for (int nd = 0; nd < 2; ++nd) {
    float aggv = 0.f;
    for (int j = 0; j < KNEI; ++j)
      aggv += wselLds[nd][j] * (float)tileH[nd][j][tid];
    out[(size_t)(n0 + nd) * DOUT + tid] = fmaxf(stLds[nd][tid] + aggv, 0.f);
  }
}

// ---------------------------------------------------------------------------
extern "C" void kernel_launch(void* const* d_in, const int* in_sizes, int n_in,
                              void* d_out, int out_size, void* d_ws, size_t ws_size,
                              hipStream_t stream) {
  const float* self_vecs  = (const float*)d_in[0];
  const float* neigh_vecs = (const float*)d_in[1];
  const float* self_w     = (const float*)d_in[2];
  const float* neigh_w    = (const float*)d_in[3];
  const float* attn       = (const float*)d_in[4];

  char* ws = (char*)d_ws;
  _Float16* Wsp = (_Float16*)ws;               // 131072 B packed self weights
  _Float16* Wnp = (_Float16*)(ws + 131072);    // 131072 B packed neigh weights
  float*    st  = (float*)(ws + 262144);       // 20000*256*4 B

  pack_weights<<<128, 512, 0, stream>>>(self_w, Wsp);
  pack_weights<<<128, 512, 0, stream>>>(neigh_w, Wnp);
  st_gemm   <<<N_NODES / 16, 128, 0, stream>>>(self_vecs, Wsp, st);
  agg_kernel<<<N_NODES / 2, 256, 0, stream>>>(neigh_vecs, Wnp, st, attn, (float*)d_out);
}
